// DiffusionOrderingNetwork_87196426043788
// MI455X (gfx1250) — compile-verified, matrix operand order checked by reading, not run on hardware
//
#include <hip/hip_runtime.h>
#include <hip/hip_bf16.h>
#include <math.h>

// ---------------------------------------------------------------------------
// CDNA5 (gfx1250) WMMA types
// ---------------------------------------------------------------------------
typedef __attribute__((ext_vector_type(16))) __bf16 v16bf;
typedef __attribute__((ext_vector_type(8)))  __bf16 v8bf;
typedef __attribute__((ext_vector_type(4)))  __bf16 v4bf;
typedef __attribute__((ext_vector_type(2)))  __bf16 v2bf;
typedef __attribute__((ext_vector_type(8)))  float  v8f;

union FragBF {
  v16bf v;
  v8bf  h[2];
};

#define HID   128
#define HEADS 4
#define NB    16
#define NN    512
#define NROWS (NB * NN)        // 8192
#define BIG   (HEADS * HID)    // 512
#define LDP   40               // LDS row stride (bf16): 80B, 16B-aligned, 20 dwords

__device__ __forceinline__ float lrelu02(float z) { return z >= 0.f ? z : 0.2f * z; }

// ---------------------------------------------------------------------------
// Generic bf16-compute WMMA GEMM:  C[M,N] = A[M,K] @ B[K,N] (+bias) (opt ReLU)
// 256 threads = 8 waves; block tile 128x64, K-step 32.
// Wave w owns rows [w*16, +16) and all 64 cols -> 4 accumulators, 4 WMMA/iter.
// A tile stored [m][k]; B tile stored TRANSPOSED [n][k] so that both A and B
// fragments are two contiguous 16B chunks per lane -> ds_load_b128.
// ---------------------------------------------------------------------------
template <bool RELU>
__global__ void gemm_bf16_wmma(const float* __restrict__ A,
                               const float* __restrict__ B,
                               const float* __restrict__ bias,
                               float* __restrict__ C,
                               int M, int K, int N) {
  __shared__ __align__(16) __bf16 As[128][LDP];   // [m][k]
  __shared__ __align__(16) __bf16 Bst[64][LDP];   // [n][k] (transposed tile)

  const int tid  = threadIdx.x;
  const int wav  = tid >> 5;      // 0..7 -> 16-row slice
  const int lane = tid & 31;
  const int hgrp = lane >> 4;     // lane half-group (0/1)
  const int mr   = lane & 15;

  const int bn = blockIdx.x * 64;
  const int bm = blockIdx.y * 128;

  v8f acc0 = {}, acc1 = {}, acc2 = {}, acc3 = {};

  for (int k0 = 0; k0 < K; k0 += 32) {
    // stage A tile 128x32 (fp32 -> bf16): float4 loads, packed b64 LDS stores
#pragma unroll
    for (int ii = 0; ii < 4; ++ii) {
      int i = ii * 256 + tid;           // 1024 quads
      int r  = i >> 3;
      int cc = (i & 7) << 2;
      const float4 av = *(const float4*)&A[(size_t)(bm + r) * K + k0 + cc];
      v4bf p;
      p[0] = (__bf16)av.x; p[1] = (__bf16)av.y;
      p[2] = (__bf16)av.z; p[3] = (__bf16)av.w;
      *(v4bf*)&As[r][cc] = p;
      if (k0 + 32 < K)
        __builtin_prefetch(&A[(size_t)(bm + r) * K + k0 + 32 + cc], 0, 0);
    }
    // stage B tile 32x64 transposed into [n][k]: 4 k-rows per thread,
    // coalesced b32 global reads across n, packed b64 LDS store
#pragma unroll
    for (int ii = 0; ii < 2; ++ii) {
      int i = ii * 256 + tid;           // 512 quads
      int n  = i & 63;
      int kb = (i >> 6) << 2;
      v4bf p;
#pragma unroll
      for (int q = 0; q < 4; ++q)
        p[q] = (__bf16)B[(size_t)(k0 + kb + q) * N + bn + n];
      *(v4bf*)&Bst[n][kb] = p;
      if (k0 + 32 < K)
        __builtin_prefetch(&B[(size_t)(k0 + 32 + kb) * N + bn + n], 0, 0);
    }
    __syncthreads();

    // A fragment: lane (hgrp,mr) -> row wav*16+mr, k-chunks {hgrp*8, 16+hgrp*8}
    FragBF af, bf0, bf1, bf2, bf3;
    const int arow = wav * 16 + mr;
    af.h[0] = *(const v8bf*)&As[arow][hgrp * 8];
    af.h[1] = *(const v8bf*)&As[arow][16 + hgrp * 8];
    bf0.h[0] = *(const v8bf*)&Bst[mr][hgrp * 8];
    bf0.h[1] = *(const v8bf*)&Bst[mr][16 + hgrp * 8];
    bf1.h[0] = *(const v8bf*)&Bst[16 + mr][hgrp * 8];
    bf1.h[1] = *(const v8bf*)&Bst[16 + mr][16 + hgrp * 8];
    bf2.h[0] = *(const v8bf*)&Bst[32 + mr][hgrp * 8];
    bf2.h[1] = *(const v8bf*)&Bst[32 + mr][16 + hgrp * 8];
    bf3.h[0] = *(const v8bf*)&Bst[48 + mr][hgrp * 8];
    bf3.h[1] = *(const v8bf*)&Bst[48 + mr][16 + hgrp * 8];

    acc0 = __builtin_amdgcn_wmma_f32_16x16x32_bf16(false, af.v, false, bf0.v,
                                                   (short)0, acc0, false, false);
    acc1 = __builtin_amdgcn_wmma_f32_16x16x32_bf16(false, af.v, false, bf1.v,
                                                   (short)0, acc1, false, false);
    acc2 = __builtin_amdgcn_wmma_f32_16x16x32_bf16(false, af.v, false, bf2.v,
                                                   (short)0, acc2, false, false);
    acc3 = __builtin_amdgcn_wmma_f32_16x16x32_bf16(false, af.v, false, bf3.v,
                                                   (short)0, acc3, false, false);
    __syncthreads();
  }

  const int cmb = bm + wav * 16 + hgrp * 8;
  v8f* accs[4] = { &acc0, &acc1, &acc2, &acc3 };
#pragma unroll
  for (int g = 0; g < 4; ++g) {
    const int cn = bn + g * 16 + mr;
    const float bb = bias ? bias[cn] : 0.f;
#pragma unroll
    for (int r = 0; r < 8; ++r) {
      float v = (*accs[g])[r] + bb;
      if (RELU) v = fmaxf(v, 0.f);
      C[(size_t)(cmb + r) * N + cn] = v;
    }
  }
}

// ---------------------------------------------------------------------------
// Row LayerNorm over D dims (+optional ReLU). One block per row.
// ---------------------------------------------------------------------------
template <int D, bool RELU>
__global__ void layernorm_kernel(const float* __restrict__ in,
                                 const float* __restrict__ w,
                                 const float* __restrict__ b,
                                 float* __restrict__ out) {
  const int row = blockIdx.x;
  const float* x = in + (size_t)row * D;
  float s = 0.f, s2 = 0.f;
  for (int c = threadIdx.x; c < D; c += blockDim.x) {
    float v = x[c];
    s += v; s2 += v * v;
  }
#pragma unroll
  for (int off = 16; off > 0; off >>= 1) {
    s  += __shfl_down(s, off, 32);
    s2 += __shfl_down(s2, off, 32);
  }
  __shared__ float shs[16], shs2[16];
  const int nw = blockDim.x >> 5;
  if ((threadIdx.x & 31) == 0) { shs[threadIdx.x >> 5] = s; shs2[threadIdx.x >> 5] = s2; }
  __syncthreads();
  if (threadIdx.x == 0) {
    float ts = shs[0], ts2 = shs2[0];
    for (int i = 1; i < nw; ++i) { ts += shs[i]; ts2 += shs2[i]; }
    shs[0] = ts; shs2[0] = ts2;
  }
  __syncthreads();
  const float mean = shs[0] * (1.0f / D);
  const float var  = shs2[0] * (1.0f / D) - mean * mean;
  const float inv  = rsqrtf(var + 1e-5f);
  for (int c = threadIdx.x; c < D; c += blockDim.x) {
    float v = (x[c] - mean) * inv * w[c] + b[c];
    if (RELU) v = fmaxf(v, 0.f);
    out[(size_t)row * D + c] = v;
  }
}

// ---------------------------------------------------------------------------
// Time embedding: sinusoid -> Linear -> LN -> SiLU -> Linear. 1 block / batch.
// ---------------------------------------------------------------------------
__global__ void time_embed_kernel(const int* __restrict__ t,
                                  const float* __restrict__ w1, const float* __restrict__ b1,
                                  const float* __restrict__ lnw, const float* __restrict__ lnb,
                                  const float* __restrict__ w2, const float* __restrict__ b2,
                                  float* __restrict__ temb) {
  const int bidx = blockIdx.x;
  const int c = threadIdx.x;   // 128 threads
  __shared__ float e[HID];
  __shared__ float shs[4], shs2[4];
  const float tv = (float)t[bidx];
  const int i = (c < 64) ? c : (c - 64);
  const float freq = __expf(-9.210340371976184f * (float)i / 64.f);
  e[c] = (c < 64) ? __cosf(tv * freq) : __sinf(tv * freq);
  __syncthreads();
  float acc = b1[c];
  for (int k = 0; k < HID; ++k) acc += e[k] * w1[k * HID + c];
  // LN across the 128 values of acc
  float s = acc, s2 = acc * acc;
#pragma unroll
  for (int off = 16; off > 0; off >>= 1) {
    s  += __shfl_down(s, off, 32);
    s2 += __shfl_down(s2, off, 32);
  }
  if ((c & 31) == 0) { shs[c >> 5] = s; shs2[c >> 5] = s2; }
  __syncthreads();
  const float ts  = shs[0] + shs[1] + shs[2] + shs[3];
  const float ts2 = shs2[0] + shs2[1] + shs2[2] + shs2[3];
  const float mean = ts * (1.0f / HID);
  const float var  = ts2 * (1.0f / HID) - mean * mean;
  const float inv  = rsqrtf(var + 1e-5f);
  float h = (acc - mean) * inv * lnw[c] + lnb[c];
  h = h / (1.f + __expf(-h));             // SiLU
  __syncthreads();
  e[c] = h;
  __syncthreads();
  float acc2 = b2[c];
  for (int k = 0; k < HID; ++k) acc2 += e[k] * w2[k * HID + c];
  temb[bidx * HID + c] = acc2;
}

// h[8192,128] += temb[row/512, c]
__global__ void add_temb_kernel(float* __restrict__ h, const float* __restrict__ temb) {
  size_t idx = (size_t)blockIdx.x * 256 + threadIdx.x;
  int row = (int)(idx >> 7);
  int c   = (int)(idx & 127);
  h[idx] += temb[(row >> 9) * HID + c];
}

// es/ed: per (row, head) dots with a_src/a_dst. One wave per (row,head).
__global__ void attn_scores_kernel(const float* __restrict__ x,
                                   const float* __restrict__ a_src,
                                   const float* __restrict__ a_dst,
                                   float* __restrict__ es, float* __restrict__ ed) {
  const int gw   = (int)((blockIdx.x * blockDim.x + threadIdx.x) >> 5);
  const int lane = threadIdx.x & 31;
  const int r  = gw >> 2;
  const int hd = gw & 3;
  const float* xp = x + (size_t)r * BIG + hd * HID;
  const float* as = a_src + hd * HID;
  const float* ad = a_dst + hd * HID;
  float ss = 0.f, sd = 0.f;
  for (int c = lane; c < HID; c += 32) {
    float v = xp[c];
    ss += v * as[c];
    sd += v * ad[c];
  }
#pragma unroll
  for (int off = 16; off > 0; off >>= 1) {
    ss += __shfl_down(ss, off, 32);
    sd += __shfl_down(sd, off, 32);
  }
  if (lane == 0) { es[r * 4 + hd] = ss; ed[r * 4 + hd] = sd; }
}

// max over j of es for each (b, head). 64 blocks.
__global__ void max_es_kernel(const float* __restrict__ es, float* __restrict__ mx) {
  const int b = blockIdx.x >> 2, hd = blockIdx.x & 3;
  float m = -1e30f;
  for (int j = threadIdx.x; j < NN; j += 256)
    m = fmaxf(m, es[((size_t)b * NN + j) * 4 + hd]);
#pragma unroll
  for (int off = 16; off > 0; off >>= 1) m = fmaxf(m, __shfl_down(m, off, 32));
  __shared__ float sh[8];
  if ((threadIdx.x & 31) == 0) sh[threadIdx.x >> 5] = m;
  __syncthreads();
  if (threadIdx.x == 0) {
    float mm = sh[0];
    for (int i = 1; i < 8; ++i) mm = fmaxf(mm, sh[i]);
    mx[blockIdx.x] = mm;
  }
}

// ---------------------------------------------------------------------------
// Fused attention aggregate: out[b, i, hd*128+c] = sum_j alpha * x * rinv + bias
// Block = (b, hd, 16-row i tile); 8 waves, wave w -> 16-col slice of 128.
// alpha generated in-register (exp of leaky-relu, max-shifted), packed to bf16
// LDS [i][j]; each thread owns a fixed alpha row so the softmax denominator is
// accumulated in registers and reduced once with width-16 shuffles; the
// RECIPROCAL is stored (16 divides/block) so the epilogue is multiply-only.
// x tile staged TRANSPOSED [c][j] so both fragments are contiguous b128 loads.
// ---------------------------------------------------------------------------
__global__ void attn_agg_kernel(const float* __restrict__ xin,
                                const float* __restrict__ es,
                                const float* __restrict__ ed,
                                const float* __restrict__ mxs,
                                const float* __restrict__ bias,
                                float* __restrict__ outp) {
  const int it = blockIdx.x & 31;           // i tile (16 rows each)
  const int hd = (blockIdx.x >> 5) & 3;     // head
  const int b  = blockIdx.x >> 7;           // batch
  const int i0 = it * 16;

  __shared__ __align__(16) __bf16 Al[16][LDP];    // alpha tile [i][j]
  __shared__ __align__(16) __bf16 Xst[128][LDP];  // x tile transposed [c][j]
  __shared__ float mrow[16], edl[16], rinv[16];

  const int tid  = threadIdx.x;
  const int wav  = tid >> 5;
  const int lane = tid & 31;
  const int hgrp = lane >> 4;
  const int mr   = lane & 15;
  const int air  = tid >> 4;                // this thread's fixed alpha row
  const int ajj  = (tid & 15) * 2;          // this thread's fixed j pair

  if (tid < 16) {
    float edv = ed[((size_t)b * NN + i0 + tid) * 4 + hd];
    edl[tid]  = edv;
    mrow[tid] = lrelu02(edv + mxs[b * 4 + hd]);
  }
  __syncthreads();

  v8f acc = {};
  float rs = 0.f;
  const float edr = edl[air];
  const float mr_ = mrow[air];

  for (int j0 = 0; j0 < NN; j0 += 32) {
    // stage alpha tile: 2 weights per thread, packed b32 store
    {
      float z0 = lrelu02(edr + es[((size_t)b * NN + j0 + ajj) * 4 + hd]);
      float z1 = lrelu02(edr + es[((size_t)b * NN + j0 + ajj + 1) * 4 + hd]);
      float w0 = __expf(z0 - mr_);
      float w1 = __expf(z1 - mr_);
      rs += w0 + w1;
      v2bf p; p[0] = (__bf16)w0; p[1] = (__bf16)w1;
      *(v2bf*)&Al[air][ajj] = p;
    }
    // stage x tile transposed [c][j]: 4 consecutive j per thread per iter,
    // coalesced b32 global reads across c, packed b64 LDS store
#pragma unroll
    for (int ii = 0; ii < 4; ++ii) {
      int i = ii * 256 + tid;               // 1024 quads
      int c  = i & 127;
      int jb = (i >> 7) << 2;
      v4bf p;
#pragma unroll
      for (int q = 0; q < 4; ++q)
        p[q] = (__bf16)xin[((size_t)b * NN + j0 + jb + q) * BIG + hd * HID + c];
      *(v4bf*)&Xst[c][jb] = p;
    }
    __syncthreads();

    FragBF af, bf;
    af.h[0] = *(const v8bf*)&Al[mr][hgrp * 8];
    af.h[1] = *(const v8bf*)&Al[mr][16 + hgrp * 8];
    const int xrow = wav * 16 + mr;
    bf.h[0] = *(const v8bf*)&Xst[xrow][hgrp * 8];
    bf.h[1] = *(const v8bf*)&Xst[xrow][16 + hgrp * 8];
    acc = __builtin_amdgcn_wmma_f32_16x16x32_bf16(false, af.v, false, bf.v,
                                                  (short)0, acc, false, false);
    __syncthreads();
  }

  // softmax denominators: reduce within each 16-lane group (one alpha row each)
  // and store the RECIPROCAL so the epilogue is multiply-only.
#pragma unroll
  for (int off = 8; off > 0; off >>= 1) rs += __shfl_down(rs, off, 16);
  if (mr == 0) rinv[wav * 2 + hgrp] = 1.0f / rs;
  __syncthreads();

  const int cg = hd * HID + wav * 16 + mr;  // global channel within 512
  const float bb = bias[cg];
#pragma unroll
  for (int r = 0; r < 8; ++r) {
    int m = i0 + hgrp * 8 + r;
    float v = acc[r] * rinv[hgrp * 8 + r] + bb;
    outp[((size_t)b * NN + m) * BIG + cg] = v;
  }
}

// final score: out[row] = mask ? dot(s_row, w2) + b2 : -inf. One wave per row.
__global__ void score_out_kernel(const float* __restrict__ s,
                                 const float* __restrict__ w2,
                                 const float* __restrict__ b2,
                                 const unsigned char* __restrict__ mask,
                                 float* __restrict__ out) {
  const int gw   = (int)((blockIdx.x * blockDim.x + threadIdx.x) >> 5);
  const int lane = threadIdx.x & 31;
  const float* sp = s + (size_t)gw * HID;
  float acc = 0.f;
  for (int c = lane; c < HID; c += 32) acc += sp[c] * w2[c];
#pragma unroll
  for (int off = 16; off > 0; off >>= 1) acc += __shfl_down(acc, off, 32);
  if (lane == 0)
    out[gw] = mask[gw] ? (acc + b2[0]) : -__builtin_inff();
}

// ---------------------------------------------------------------------------
// Host launcher. Input order: x, edge_index, edge_attr, mask, t, then params
// pytree leaves (dict keys sorted): gat[0..3]{W,a_dst,a_src,bias,ln_b,ln_w},
// node_embed{b1,b2,ln_b,ln_w,w1,w2}, score{b1,b2,w1,w2}, time_embed{...}.
// ---------------------------------------------------------------------------
extern "C" void kernel_launch(void* const* d_in, const int* in_sizes, int n_in,
                              void* d_out, int out_size, void* d_ws, size_t ws_size,
                              hipStream_t stream) {
  const float* x            = (const float*)d_in[0];
  const unsigned char* mask = (const unsigned char*)d_in[3];
  const int* t              = (const int*)d_in[4];

  struct GatP { const float *W, *a_dst, *a_src, *bias, *ln_b, *ln_w; };
  GatP gat[4];
  for (int l = 0; l < 4; ++l) {
    int k = 5 + l * 6;
    gat[l].W     = (const float*)d_in[k + 0];
    gat[l].a_dst = (const float*)d_in[k + 1];
    gat[l].a_src = (const float*)d_in[k + 2];
    gat[l].bias  = (const float*)d_in[k + 3];
    gat[l].ln_b  = (const float*)d_in[k + 4];
    gat[l].ln_w  = (const float*)d_in[k + 5];
  }
  const float* ne_b1  = (const float*)d_in[29];
  const float* ne_b2  = (const float*)d_in[30];
  const float* ne_lnb = (const float*)d_in[31];
  const float* ne_lnw = (const float*)d_in[32];
  const float* ne_w1  = (const float*)d_in[33];
  const float* ne_w2  = (const float*)d_in[34];
  const float* sc_b1  = (const float*)d_in[35];
  const float* sc_b2  = (const float*)d_in[36];
  const float* sc_w1  = (const float*)d_in[37];
  const float* sc_w2  = (const float*)d_in[38];
  const float* te_b1  = (const float*)d_in[39];
  const float* te_b2  = (const float*)d_in[40];
  const float* te_lnb = (const float*)d_in[41];
  const float* te_lnw = (const float*)d_in[42];
  const float* te_w1  = (const float*)d_in[43];
  const float* te_w2  = (const float*)d_in[44];

  char* ws = (char*)d_ws;
  const size_t MB = 1ull << 20;
  float* bufX  = (float*)(ws + 0 * MB);    // [8192,512] h@W
  float* bufO  = (float*)(ws + 16 * MB);   // [8192,512] attention out
  float* bufH  = (float*)(ws + 32 * MB);   // [8192,512] current hidden
  float* bufT  = (float*)(ws + 48 * MB);   // [8192,128] tmp
  float* temb  = (float*)(ws + 52 * MB);           // [16,128]
  float* esb   = (float*)(ws + 52 * MB + 65536);   // [8192,4]
  float* edb   = (float*)(ws + 52 * MB + 65536 + 131072);
  float* mxs   = (float*)(ws + 52 * MB + 65536 + 262144);

  // 1) time embedding
  time_embed_kernel<<<NB, 128, 0, stream>>>(t, te_w1, te_b1, te_lnw, te_lnb,
                                            te_w2, te_b2, temb);

  // 2) node embed MLP
  gemm_bf16_wmma<false><<<dim3(HID / 64, NROWS / 128), 256, 0, stream>>>(
      x, ne_w1, ne_b1, bufT, NROWS, 64, HID);
  layernorm_kernel<HID, true><<<NROWS, 128, 0, stream>>>(bufT, ne_lnw, ne_lnb, bufT);
  gemm_bf16_wmma<false><<<dim3(HID / 64, NROWS / 128), 256, 0, stream>>>(
      bufT, ne_w2, ne_b2, bufH, NROWS, HID, HID);
  add_temb_kernel<<<(NROWS * HID) / 256, 256, 0, stream>>>(bufH, temb);

  // 3) GAT layers
  for (int l = 0; l < 4; ++l) {
    const int din = (l == 0) ? HID : BIG;
    gemm_bf16_wmma<false><<<dim3(BIG / 64, NROWS / 128), 256, 0, stream>>>(
        bufH, gat[l].W, nullptr, bufX, NROWS, din, BIG);
    attn_scores_kernel<<<(NROWS * HEADS) / 8, 256, 0, stream>>>(
        bufX, gat[l].a_src, gat[l].a_dst, esb, edb);
    max_es_kernel<<<NB * HEADS, 256, 0, stream>>>(esb, mxs);
    attn_agg_kernel<<<NB * HEADS * (NN / 16), 256, 0, stream>>>(
        bufX, esb, edb, mxs, gat[l].bias, bufO);
    layernorm_kernel<BIG, true><<<NROWS, 256, 0, stream>>>(
        bufO, gat[l].ln_w, gat[l].ln_b, bufH);
  }

  // 4) score head
  gemm_bf16_wmma<true><<<dim3(HID / 64, NROWS / 128), 256, 0, stream>>>(
      bufH, sc_w1, sc_b1, bufT, NROWS, BIG, HID);
  score_out_kernel<<<NROWS / 8, 256, 0, stream>>>(bufT, sc_w2, sc_b2, mask,
                                                  (float*)d_out);
}
